// DogDetector_36094905155657
// MI455X (gfx1250) — matrix operand order repaired and checked by reading, not compiled, not run on hardware
//
#include <hip/hip_runtime.h>
#include <hip/hip_bf16.h>

typedef __attribute__((ext_vector_type(16))) _Float16 v16h;
typedef __attribute__((ext_vector_type(8)))  _Float16 v8h;
typedef __attribute__((ext_vector_type(8)))  float    v8f;
typedef __attribute__((ext_vector_type(4)))  float    v4f;

#define NEG_INF (-__builtin_huge_valf())

// ---------------------------------------------------------------------------
// Zero a byte range (16B granules). Used once per call to clear the padded
// activation region (zero halos for boundary-check-free conv loads).
// ---------------------------------------------------------------------------
__global__ void zero_bytes(v4f* __restrict__ p, long long n16) {
  long long i = (long long)blockIdx.x * blockDim.x + threadIdx.x;
  if (i < n16) p[i] = (v4f){0.f, 0.f, 0.f, 0.f};
}

// ---------------------------------------------------------------------------
// Weight repack. K order is tap-major: k = (kh*3+kw)*Cin + ci (a 32-wide K
// chunk = 32 consecutive channels of ONE tap -> contiguous NHWC loads).
// A-fragment lane layout used by this code (B loads use the same k order):
//   lanes 0-15  (row M=lane)    hold K offsets {0..15}
//   lanes 16-31 (row M=lane-16) hold K offsets {16..31}
// Packed as [coTile][kTile][lane][16] contiguous halves (32B/lane, aligned).
// ---------------------------------------------------------------------------
__global__ void pack_w(const float* __restrict__ W, _Float16* __restrict__ A,
                       int Cout, int Cin, int coTiles) {
  int K = Cin * 9;
  int kTiles = K / 32;
  int total = coTiles * kTiles * 32 * 16;
  for (int idx = blockIdx.x * blockDim.x + threadIdx.x; idx < total;
       idx += gridDim.x * blockDim.x) {
    int e    = idx & 15;
    int lane = (idx >> 4) & 31;
    int kt   = (idx >> 9) % kTiles;
    int ct   = (idx >> 9) / kTiles;
    int m    = lane & 15;
    int koff = e + ((lane >> 4) << 4);
    int co   = ct * 16 + m;
    int k    = kt * 32 + koff;
    int tap  = k / Cin;                  // kh*3+kw
    int ci   = k - tap * Cin;
    float v  = (co < Cout) ? W[((size_t)co * Cin + ci) * 9 + tap] : 0.0f;
    A[idx] = (_Float16)v;
  }
}

// ---------------------------------------------------------------------------
// Stem: 7x7/s2 pad3 conv (3->64) + ReLU + 2x2 maxpool, fused.
// x: [16][3][512][512] fp32 -> interior of padded NHWC f16 [16][130][130][64].
// Consecutive threads = consecutive channels of one pixel (broadcast window).
// ---------------------------------------------------------------------------
__global__ void stem_pool(const float* __restrict__ x,
                          const float* __restrict__ w,
                          _Float16* __restrict__ out) {
  int idx = blockIdx.x * blockDim.x + threadIdx.x;   // 16*128*128*64 threads
  int c  = idx & 63;
  int px = (idx >> 6) & 127;
  int py = (idx >> 13) & 127;
  int n  = idx >> 20;
  const float* xN = x + (size_t)n * 3 * 512 * 512;
  const float* wc = w + c * 147;                     // [3][7][7]
  float m = NEG_INF;
#pragma unroll
  for (int sy = 0; sy < 2; ++sy) {
#pragma unroll
    for (int sx = 0; sx < 2; ++sx) {
      int oy = 2 * py + sy, ox = 2 * px + sx;        // pre-pool 256x256 coords
      float acc = 0.0f;
      for (int ci = 0; ci < 3; ++ci) {
#pragma unroll
        for (int kh = 0; kh < 7; ++kh) {
          int hi = oy * 2 - 3 + kh;
          if ((unsigned)hi >= 512u) continue;
          const float* row = xN + ((size_t)ci * 512 + hi) * 512;
          const float* wr  = wc + (ci * 7 + kh) * 7;
#pragma unroll
          for (int kw = 0; kw < 7; ++kw) {
            int wi = ox * 2 - 3 + kw;
            if ((unsigned)wi < 512u) acc = fmaf(row[wi], wr[kw], acc);
          }
        }
      }
      m = fmaxf(m, acc);
    }
  }
  size_t o = (((size_t)n * 130 + py + 1) * 130 + px + 1) * 64 + c;  // halo +1
  out[o] = (_Float16)fmaxf(m, 0.0f);                 // relu(max)==max(relu)
}

// ---------------------------------------------------------------------------
// 3x3 conv as implicit GEMM on WMMA, NHWC activations with 1-pixel zero halo
// (no boundary checks).  One wave32 computes TWO 16(Cout) x 16(pos) tiles
// sharing each A fragment: inner loop = 1 A load + 2 B loads + 2 wmma.
// in : padded NHWC f16 [N][HinP][WinP][Cin]   A : packed weights (see pack_w)
// out: f16 -> padded NHWC (halo interior, vector store); fp32 -> unpadded.
// ---------------------------------------------------------------------------
template <bool RELU, bool OUT16>
__global__ void conv3x3_wmma(const _Float16* __restrict__ in,
                             const _Float16* __restrict__ A,
                             _Float16* __restrict__ out16,
                             float* __restrict__ out32,
                             int Cin, int Cout, int coTiles,
                             int HinP, int WinP,
                             int Hout, int Wout,
                             int stride, int Nimg) {
  const int chunkCnt = Cin >> 5;                 // 32-channel chunks per tap
  const int kTiles   = 9 * chunkCnt;
  const int posPairs = (Hout * Wout) >> 5;       // 2 pos-tiles per wave
  const int lane = threadIdx.x & 31;

  int tile = blockIdx.x * (blockDim.x >> 5) + (threadIdx.x >> 5);
  int total = Nimg * coTiles * posPairs;
  if (tile >= total) return;                     // wave-uniform exit

  int pp = tile % posPairs;
  int t2 = tile / posPairs;
  int ct = t2 % coTiles;
  int n  = t2 / coTiles;

  int pos0 = pp * 32 + (lane & 15);
  int pos1 = pos0 + 16;
  int ho0 = pos0 / Wout, wo0 = pos0 % Wout;      // Wout is a power of two
  int ho1 = pos1 / Wout, wo1 = pos1 % Wout;
  int khi = (lane & 16);                         // B frag: +16 ci, upper half

  const _Float16* inN = in + (size_t)n * HinP * WinP * Cin;
  const v16h* ap = (const v16h*)A + (size_t)ct * kTiles * 32 + lane;

  v8f c0 = {}, c1 = {};
  int kt = 0;
#pragma unroll
  for (int kh = 0; kh < 3; ++kh) {
#pragma unroll
    for (int kw = 0; kw < 3; ++kw) {
      // halo shift: padded row = ho*stride - 1 + kh + 1 = ho*stride + kh >= 0
      size_t b0 = ((size_t)(ho0 * stride + kh) * WinP + (wo0 * stride + kw)) * Cin + khi;
      size_t b1 = ((size_t)(ho1 * stride + kh) * WinP + (wo1 * stride + kw)) * Cin + khi;
      for (int cc = 0; cc < chunkCnt; ++cc, ++kt) {
        v16h a  = ap[(size_t)kt * 32];
        v16h x0 = *(const v16h*)(inN + b0 + cc * 32);
        v16h x1 = *(const v16h*)(inN + b1 + cc * 32);
        c0 = __builtin_amdgcn_wmma_f32_16x16x32_f16(false, a, false, x0,
                                                    (short)0, c0, false, false);
        c1 = __builtin_amdgcn_wmma_f32_16x16x32_f16(false, a, false, x1,
                                                    (short)0, c1, false, false);
      }
    }
  }

  // C/D layout: VGPR r -> M = r + (lane>=16 ? 8 : 0), N = lane%16.
  // NHWC: lane's 8 output channels are contiguous -> one 16B vector store.
  int mBase = (lane >> 4) << 3;
#pragma unroll
  for (int t = 0; t < 2; ++t) {
    int ho = t ? ho1 : ho0;
    int wo = t ? wo1 : wo0;
    v8f& c = t ? c1 : c0;
    if (OUT16) {
      size_t p = (((size_t)n * (Hout + 2) + ho + 1) * (Wout + 2) + wo + 1) * Cout
               + ct * 16 + mBase;
      v8h o;
#pragma unroll
      for (int r = 0; r < 8; ++r) {
        float v = c[r];
        if (RELU) v = fmaxf(v, 0.0f);
        o[r] = (_Float16)v;
      }
      *(v8h*)(out16 + p) = o;
    } else {
      size_t p = (((size_t)n * Hout + ho) * Wout + wo) * Cout + ct * 16 + mBase;
#pragma unroll
      for (int r = 0; r < 8; ++r) {
        int co = ct * 16 + mBase + r;
        if (co < Cout) {
          float v = c[r];
          if (RELU) v = fmaxf(v, 0.0f);
          out32[p + r] = v;
        }
      }
    }
  }
}

// ---------------------------------------------------------------------------
// Anchor decode + sigmoid + clip + sequential NMS (100 rounds), one WGP per
// image; all 2304 candidates in LDS (~56KB of 320KB) so NMS rounds are
// LDS-only.  bbox/conf are NHWC fp32 (unpadded).  out: [16][100][5] fp32.
// ---------------------------------------------------------------------------
#define NANCH 2304
__global__ void decode_nms(const float* __restrict__ bbox,
                           const float* __restrict__ conf,
                           float* __restrict__ out) {
  __shared__ float bx1[NANCH], by1[NANCH], bx2[NANCH], by2[NANCH];
  __shared__ float sc[NANCH], s[NANCH];
  __shared__ float rv[256];
  __shared__ int   ri[256];

  const int b = blockIdx.x;
  const int tid = threadIdx.x;
  const float SC[3] = {0.25f, 0.5f, 0.75f};
  const float RT[3] = {0.7f, 1.0f, 1.3f};

  for (int t = tid; t < NANCH; t += 256) {
    int a = t % 9;
    int cell = t / 9;                  // = i*16 + j
    int j = cell % 16;                 // x
    int i = cell / 16;                 // y
    int si = a / 3, rj = a % 3;
    float sq = sqrtf(RT[rj]);
    float aw = SC[si] * sq, ah = SC[si] / sq;
    float acx = (j + 0.5f) * 0.0625f, acy = (i + 0.5f) * 0.0625f;
    size_t base = ((size_t)b * 256 + cell) * 36 + a * 4;     // NHWC bbox
    float p0 = bbox[base + 0], p1 = bbox[base + 1];
    float p2 = bbox[base + 2], p3 = bbox[base + 3];
    float cf = conf[((size_t)b * 256 + cell) * 9 + a];       // NHWC conf
    float sg = 1.0f / (1.0f + expf(-cf));
    float cx = acx + p0 * aw, cy = acy + p1 * ah;
    float wb = aw * expf(p2), hb = ah * expf(p3);
    float x1 = fminf(fmaxf(cx - 0.5f * wb, 0.0f), 1.0f);
    float y1 = fminf(fmaxf(cy - 0.5f * hb, 0.0f), 1.0f);
    float x2 = fminf(fmaxf(cx + 0.5f * wb, 0.0f), 1.0f);
    float y2 = fminf(fmaxf(cy + 0.5f * hb, 0.0f), 1.0f);
    bx1[t] = x1; by1[t] = y1; bx2[t] = x2; by2[t] = y2;
    sc[t] = sg;
    s[t] = (sg > 0.5f) ? sg : NEG_INF;
  }
  __syncthreads();

  for (int d = 0; d < 100; ++d) {
    // argmax (first-index tie-break, matching jnp.argmax)
    float bv = NEG_INF;
    int bi = 0x7fffffff;
    for (int t = tid; t < NANCH; t += 256) {
      float v = s[t];
      if (v > bv || (v == bv && t < bi)) { bv = v; bi = t; }
    }
    rv[tid] = bv; ri[tid] = bi;
    __syncthreads();
    for (int off = 128; off > 0; off >>= 1) {
      if (tid < off) {
        float v = rv[tid + off]; int ii = ri[tid + off];
        if (v > rv[tid] || (v == rv[tid] && ii < ri[tid])) {
          rv[tid] = v; ri[tid] = ii;
        }
      }
      __syncthreads();
    }
    int   i  = ri[0];
    float sv = rv[0];
    __syncthreads();

    bool valid = sv > 0.5f;
    float ix1 = bx1[i], iy1 = by1[i], ix2 = bx2[i], iy2 = by2[i];
    float a1 = (ix2 - ix1) * (iy2 - iy1);
    if (valid) {
      for (int t = tid; t < NANCH; t += 256) {
        float xx1 = fmaxf(ix1, bx1[t]);
        float yy1 = fmaxf(iy1, by1[t]);
        float xx2 = fminf(ix2, bx2[t]);
        float yy2 = fminf(iy2, by2[t]);
        float inter = fmaxf(xx2 - xx1, 0.0f) * fmaxf(yy2 - yy1, 0.0f);
        float a2 = (bx2[t] - bx1[t]) * (by2[t] - by1[t]);
        float iou = inter / (a1 + a2 - inter + 1e-9f);
        if (iou > 0.5f) s[t] = NEG_INF;
      }
    }
    if (tid == 0) {
      s[i] = NEG_INF;
      float* o = out + ((size_t)b * 100 + d) * 5;
      o[0] = valid ? ix1 : 0.0f;
      o[1] = valid ? iy1 : 0.0f;
      o[2] = valid ? ix2 : 0.0f;
      o[3] = valid ? iy2 : 0.0f;
      o[4] = valid ? sc[i] : 0.0f;
    }
    __syncthreads();
  }
}

// ---------------------------------------------------------------------------
extern "C" void kernel_launch(void* const* d_in, const int* in_sizes, int n_in,
                              void* d_out, int out_size, void* d_ws,
                              size_t ws_size, hipStream_t stream) {
  (void)in_sizes; (void)n_in; (void)out_size; (void)ws_size;
  const float* x      = (const float*)d_in[0];
  const float* w_stem = (const float*)d_in[1];
  const float* w2     = (const float*)d_in[2];
  const float* w3     = (const float*)d_in[3];
  const float* w4     = (const float*)d_in[4];
  const float* w_head = (const float*)d_in[5];
  const float* w_box  = (const float*)d_in[6];
  const float* w_conf = (const float*)d_in[7];

  char* ws = (char*)d_ws;
  size_t off = 0;
  auto alloc = [&](size_t bytes) -> char* {
    char* p = ws + off;
    off = (off + bytes + 255) & ~(size_t)255;
    return p;
  };

  // Padded NHWC activations (1-pixel zero halo), contiguous region first so
  // one zero pass clears every halo.
  _Float16* f1   = (_Float16*)alloc((size_t)16 * 130 * 130 * 64 * 2);
  _Float16* f2   = (_Float16*)alloc((size_t)16 * 66 * 66 * 128 * 2);
  _Float16* f3   = (_Float16*)alloc((size_t)16 * 34 * 34 * 256 * 2);
  _Float16* f4   = (_Float16*)alloc((size_t)16 * 18 * 18 * 512 * 2);
  _Float16* hh   = (_Float16*)alloc((size_t)16 * 18 * 18 * 256 * 2);
  size_t zeroBytes = off;                        // everything above gets zeroed

  float*    bboxB= (float*)   alloc((size_t)16 * 256 * 36 * 4);
  float*    confB= (float*)   alloc((size_t)16 * 256 * 9 * 4);
  _Float16* A2   = (_Float16*)alloc((size_t)8  * 18  * 512 * 2);
  _Float16* A3   = (_Float16*)alloc((size_t)16 * 36  * 512 * 2);
  _Float16* A4   = (_Float16*)alloc((size_t)32 * 72  * 512 * 2);
  _Float16* Ah   = (_Float16*)alloc((size_t)16 * 144 * 512 * 2);
  _Float16* Abox = (_Float16*)alloc((size_t)3  * 72  * 512 * 2);
  _Float16* Acf  = (_Float16*)alloc((size_t)1  * 72  * 512 * 2);

  // Clear padded activation region (halo zeros; interiors are overwritten).
  long long n16 = (long long)(zeroBytes / 16);
  zero_bytes<<<(unsigned)((n16 + 255) / 256), 256, 0, stream>>>((v4f*)ws, n16);

  auto packGrid = [](int coTiles, int kTiles) {
    return (coTiles * kTiles * 512 + 255) / 256;
  };
  pack_w<<<packGrid(8, 18),   256, 0, stream>>>(w2,     A2,   128, 64,  8);
  pack_w<<<packGrid(16, 36),  256, 0, stream>>>(w3,     A3,   256, 128, 16);
  pack_w<<<packGrid(32, 72),  256, 0, stream>>>(w4,     A4,   512, 256, 32);
  pack_w<<<packGrid(16, 144), 256, 0, stream>>>(w_head, Ah,   256, 512, 16);
  pack_w<<<packGrid(3, 72),   256, 0, stream>>>(w_box,  Abox, 36,  256, 3);
  pack_w<<<packGrid(1, 72),   256, 0, stream>>>(w_conf, Acf,  9,   256, 1);

  // Stem + ReLU + maxpool fused -> padded f1 interior
  stem_pool<<<(16 * 128 * 128 * 64) / 256, 256, 0, stream>>>(x, w_stem, f1);

  // pair-tiles = N * coTiles * (Hout*Wout/32); 8 waves per 256-thread block
  conv3x3_wmma<true, true><<<(16 * 8 * 128) / 8, 256, 0, stream>>>(
      f1, A2, f2, nullptr, 64, 128, 8, 130, 130, 64, 64, 2, 16);
  conv3x3_wmma<true, true><<<(16 * 16 * 32) / 8, 256, 0, stream>>>(
      f2, A3, f3, nullptr, 128, 256, 16, 66, 66, 32, 32, 2, 16);
  conv3x3_wmma<true, true><<<(16 * 32 * 8) / 8, 256, 0, stream>>>(
      f3, A4, f4, nullptr, 256, 512, 32, 34, 34, 16, 16, 2, 16);
  conv3x3_wmma<true, true><<<(16 * 16 * 8) / 8, 256, 0, stream>>>(
      f4, Ah, hh, nullptr, 512, 256, 16, 18, 18, 16, 16, 1, 16);
  conv3x3_wmma<false, false><<<(16 * 3 * 8) / 8, 256, 0, stream>>>(
      hh, Abox, nullptr, bboxB, 256, 36, 3, 18, 18, 16, 16, 1, 16);
  conv3x3_wmma<false, false><<<(16 * 1 * 8) / 8, 256, 0, stream>>>(
      hh, Acf, nullptr, confB, 256, 9, 1, 18, 18, 16, 16, 1, 16);

  decode_nms<<<16, 256, 0, stream>>>(bboxB, confB, (float*)d_out);
}
